// STGCN_51616916963637
// MI455X (gfx1250) — matrix operand order repaired and checked by reading, not compiled, not run on hardware
//
#include <hip/hip_runtime.h>
#include <hip/hip_bf16.h>
#include <math.h>

// ---------------------------------------------------------------------------
// STGCN fully-fused kernel for MI455X (gfx1250, wave32, WMMA).
// All stages are per-node; each wave owns 16 nodes, activations stay in LDS
// as f16, GEMMs run on v_wmma_f32_16x16x32_f16 with f32 accumulation.
// Round 2: force-inline all helpers so LDS accesses lower to ds_load/ds_store
// (round 1 emitted flat_load for A-fragments), vectorize A loads to b128,
// roll the temporal A window (1 LDS A-load per step instead of 3), and use
// v_rcp_f32 for the sigmoid instead of a full IEEE divide.
// ---------------------------------------------------------------------------

typedef __attribute__((ext_vector_type(16))) _Float16     v16h;
typedef __attribute__((ext_vector_type(8)))  float        v8f;
typedef __attribute__((ext_vector_type(4)))  unsigned int v4u;

#define N_NODES 50000
#define NPW 16                 // nodes per wave
#define WAVES 4                // waves per block
#define NPB (NPW * WAVES)      // nodes per block
#define S0 (21 * 32 + 8)       // LDS node stride (halves), buffer 0 (max T=21)
#define S1 (19 * 32 + 8)       // LDS node stride (halves), buffer 1 (max T=19)

union AV { v4u q[2]; v16h v; };

static __device__ __forceinline__ v8f wmma16(v16h a, v16h b, v8f c) {
  // D = A(16x32 f16) * B(32x16 f16) + C(16x16 f32)
  return __builtin_amdgcn_wmma_f32_16x16x32_f16(false, a, false, b, (short)0, c,
                                                false, false);
}

// A-matrix fragment: 16 nodes (rows) x 32 channels (K) at time row `trow`.
// 16-bit A layout: lane m<16 -> M=m, halves 2j+e hold K=(j>=4?16:0)+2(j&3)+e;
// lanes 16-31 same M with K offset +8 in each 16-group. Both 4-dword groups
// are 16B-aligned (strides are multiples of 4 dwords) -> two ds_load_b128.
static __device__ __forceinline__ v16h loadA(const _Float16* nb, int stride,
                                             int trow, int lane) {
  const int m = lane & 15;
  const _Float16* base = nb + m * stride + trow * 32 + ((lane & 16) ? 8 : 0);
  AV a;
  a.q[0] = *(const v4u*)base;        // K group 0..7 (+hi offset)
  a.q[1] = *(const v4u*)(base + 16); // K group 16..23 (+hi offset)
  return a.v;
}

// B-matrix fragment (32x16) for temporal conv: B[K=kt*32+c][N=o] = W[o][c][kt].
// 16-bit B layout: lanes 0-15 -> N=lane, K=0..7,16..23; lanes 16-31 K=8..15,24..31.
static __device__ __forceinline__ v16h loadBconv(const float* w, int ntile,
                                                 int kt, int lane) {
  const int n = ntile * 16 + (lane & 15);
  const int cbase = (lane & 16) >> 1; // 0 or 8
  v16h r;
#pragma unroll
  for (int h = 0; h < 16; ++h) {
    const int c = cbase + (h & 7) + ((h & 8) << 1);
    r[h] = (_Float16)w[n * 96 + c * 3 + kt]; // w: (cout=32, cin=32, 1, KT=3)
  }
  return r;
}

// Store 16x16 f32 C/D tile (already activated) to LDS as f16.
// C/D layout: VGPR r, lane l -> M = r + (l>=16?8:0), N = (l&15) + 16*ntile.
static __device__ __forceinline__ void storeD(_Float16* nb, int stride, int t,
                                              int ntile, int lane, v8f d) {
  const int hi8 = (lane & 16) ? 8 : 0;
  const int n = ntile * 16 + (lane & 15);
#pragma unroll
  for (int r = 0; r < 8; ++r)
    nb[(r + hi8) * stride + t * 32 + n] = (_Float16)d[r];
}

// Gated temporal conv: out[t] = relu(P * sigmoid(Q) + R), each a (16n x 32o)
// GEMM with K = 96 (3 temporal taps x 32 channels). Rolling A window.
static __device__ __forceinline__ void tconv(
    const _Float16* in, int sin, int TIN, _Float16* out, int sout,
    const float* wP, const float* wQ, const float* wR,
    const float* bP, const float* bQ, const float* bR, int lane) {
  const int TOUT = TIN - 2;
#pragma unroll
  for (int nt = 0; nt < 2; ++nt) {
    v16h BP[3], BQ[3], BR[3];
#pragma unroll
    for (int kt = 0; kt < 3; ++kt) {
      BP[kt] = loadBconv(wP, nt, kt, lane);
      BQ[kt] = loadBconv(wQ, nt, kt, lane);
      BR[kt] = loadBconv(wR, nt, kt, lane);
    }
    const int o = nt * 16 + (lane & 15);
    const float biasP = bP[o], biasQ = bQ[o], biasR = bR[o];
    v16h A0 = loadA(in, sin, 0, lane);
    v16h A1 = loadA(in, sin, 1, lane);
    for (int t = 0; t < TOUT; ++t) {
      v16h A2 = loadA(in, sin, t + 2, lane);
      v8f accP, accQ, accR;
#pragma unroll
      for (int r = 0; r < 8; ++r) {
        accP[r] = biasP; accQ[r] = biasQ; accR[r] = biasR;
      }
      accP = wmma16(A0, BP[0], accP);
      accQ = wmma16(A0, BQ[0], accQ);
      accR = wmma16(A0, BR[0], accR);
      accP = wmma16(A1, BP[1], accP);
      accQ = wmma16(A1, BQ[1], accQ);
      accR = wmma16(A1, BR[1], accR);
      accP = wmma16(A2, BP[2], accP);
      accQ = wmma16(A2, BQ[2], accQ);
      accR = wmma16(A2, BR[2], accR);
      v8f h;
#pragma unroll
      for (int r = 0; r < 8; ++r) {
        float sg = __builtin_amdgcn_rcpf(1.0f + __expf(-accQ[r]));
        h[r] = fmaxf(accP[r] * sg + accR[r], 0.0f);
      }
      storeD(out, sout, t, nt, lane, h);
      A0 = A1;
      A1 = A2;
    }
  }
}

// ChebConv K=1: per-time-step 32x32 linear + relu. B[K=c][N=o] = cw[o][c].
static __device__ __forceinline__ void cheb(const _Float16* in, int sin, int T,
                                            _Float16* out, int sout,
                                            const float* cw, const float* cb,
                                            int lane) {
#pragma unroll
  for (int nt = 0; nt < 2; ++nt) {
    const int n = nt * 16 + (lane & 15);
    const int cbase = (lane & 16) >> 1;
    v16h B;
#pragma unroll
    for (int h = 0; h < 16; ++h) B[h] = (_Float16)cw[n * 32 + cbase + (h & 7) + ((h & 8) << 1)];
    const float bias = cb[n];
    for (int t = 0; t < T; ++t) {
      v16h A = loadA(in, sin, t, lane);
      v8f acc;
#pragma unroll
      for (int r = 0; r < 8; ++r) acc[r] = bias;
      acc = wmma16(A, B, acc);
      v8f h;
#pragma unroll
      for (int r = 0; r < 8; ++r) h[r] = fmaxf(acc[r], 0.0f);
      storeD(out, sout, t, nt, lane, h);
    }
  }
}

// Per-node BatchNorm (training-mode batch stats over T x C) + relu, in place.
// Lane pair (l, l^16) covers one node: 16 channels each, combined via shfl_xor.
static __device__ __forceinline__ void bnRelu17(_Float16* buf, int sin,
                                                const float* bg,
                                                const float* bb,
                                                int nodeClamped, int lane) {
  const int m = lane & 15;
  const int ch0 = (lane & 16);
  _Float16* p = buf + m * sin + ch0;
  float s = 0.f, ss = 0.f;
  for (int t = 0; t < 17; ++t) {
#pragma unroll
    for (int c = 0; c < 16; ++c) {
      float v = (float)p[t * 32 + c];
      s += v; ss += v * v;
    }
  }
  s  += __shfl_xor(s, 16, 32);
  ss += __shfl_xor(ss, 16, 32);
  const float inv = 1.0f / (17.0f * 32.0f);
  const float mean = s * inv;
  const float var  = ss * inv - mean * mean;
  const float rs   = rsqrtf(var + 1e-5f);
  const float sc   = bg[nodeClamped] * rs;
  const float sh   = bb[nodeClamped] - mean * sc;
  for (int t = 0; t < 17; ++t) {
#pragma unroll
    for (int c = 0; c < 16; ++c) {
      float v = fmaxf((float)p[t * 32 + c] * sc + sh, 0.0f);
      p[t * 32 + c] = (_Float16)v;
    }
  }
}

// Final per-node BN (T=13, no relu) fused with the channel mean; accumulates
// sum over (channels, nodes) per time step into S[13] via atomics.
static __device__ __forceinline__ void finalStage(const _Float16* in, int sin,
                                                  const float* bg,
                                                  const float* bb,
                                                  int nodeClamped, bool valid,
                                                  int lane, float* S) {
  const int m = lane & 15;
  const int ch0 = (lane & 16);
  const _Float16* p = in + m * sin + ch0;
  float st[13];
  float ss = 0.f, sum = 0.f;
#pragma unroll
  for (int t = 0; t < 13; ++t) {
    float s = 0.f;
#pragma unroll
    for (int c = 0; c < 16; ++c) {
      float v = (float)p[t * 32 + c];
      s += v; ss += v * v;
    }
    st[t] = s; sum += s;
  }
#pragma unroll
  for (int t = 0; t < 13; ++t) st[t] += __shfl_xor(st[t], 16, 32);
  ss  += __shfl_xor(ss, 16, 32);
  sum += __shfl_xor(sum, 16, 32);
  const float inv = 1.0f / (13.0f * 32.0f);
  const float mean = sum * inv;
  const float var  = ss * inv - mean * mean;
  const float rs   = rsqrtf(var + 1e-5f);
  const float sc   = bg[nodeClamped] * rs;
  const float sh2  = 32.0f * (bb[nodeClamped] - sc * mean);
#pragma unroll
  for (int t = 0; t < 13; ++t) {
    float v = valid ? (sc * st[t] + sh2) : 0.0f; // sum over 32 BN'd channels
    v += __shfl_xor(v, 1, 32);
    v += __shfl_xor(v, 2, 32);
    v += __shfl_xor(v, 4, 32);
    v += __shfl_xor(v, 8, 32);   // lanes 0..15 reduced (16..31 are duplicates)
    if (lane == 0) atomicAdd(&S[t], v);
  }
}

struct STGCNParams {
  const float* x;
  const float* w[4][3];   // tconv weights: {st1.tc1, st1.tc2, st2.tc1, st2.tc2}
  const float* bw[4][3];  // tconv biases
  const float* cw[2];     // cheb weights
  const float* cb[2];     // cheb biases
  const float* bng[2];
  const float* bnb[2];
  float* S;               // 13 accumulators in d_ws
};

__global__ __launch_bounds__(WAVES * 32) void stgcn_main(STGCNParams p) {
  extern __shared__ _Float16 smem[];
  const int tid  = threadIdx.x;
  const int lane = tid & 31;
  const int wv   = tid >> 5;
  const int blockBase = blockIdx.x * NPB;

  _Float16* buf0 = smem + wv * NPW * S0;
  _Float16* buf1 = smem + WAVES * NPW * S0 + wv * NPW * S1;

  // ---- cooperative load: x (T=21, N, C=32) -> LDS f16, node-major tiles ----
  const int total = NPB * 21 * 32;
  for (int i = tid; i < total; i += WAVES * 32) {
    int nl  = i / (21 * 32);
    int rem = i - nl * (21 * 32);
    int t = rem >> 5;
    int c = rem & 31;
    int gn = blockBase + nl;
    if (gn > N_NODES - 1) gn = N_NODES - 1;
    float v = p.x[((long)t * N_NODES + gn) * 32 + c];
    smem[(nl >> 4) * NPW * S0 + (nl & 15) * S0 + t * 32 + c] = (_Float16)v;
  }
  __syncthreads();

  const int myNode = blockBase + wv * NPW + (lane & 15);
  const int nc = (myNode < N_NODES) ? myNode : (N_NODES - 1);
  const bool valid = myNode < N_NODES;

  // ---- STConv block 1 ----
  tconv(buf0, S0, 21, buf1, S1, p.w[0][0], p.w[0][1], p.w[0][2],
        p.bw[0][0], p.bw[0][1], p.bw[0][2], lane);                 // X(21) -> T0(19)
  cheb(buf1, S1, 19, buf0, S0, p.cw[0], p.cb[0], lane);            // T0 -> Tg(19)
  tconv(buf0, S0, 19, buf1, S1, p.w[1][0], p.w[1][1], p.w[1][2],
        p.bw[1][0], p.bw[1][1], p.bw[1][2], lane);                 // Tg -> Tt(17)
  bnRelu17(buf1, S1, p.bng[0], p.bnb[0], nc, lane);                // BN + top relu

  // ---- STConv block 2 ----
  tconv(buf1, S1, 17, buf0, S0, p.w[2][0], p.w[2][1], p.w[2][2],
        p.bw[2][0], p.bw[2][1], p.bw[2][2], lane);                 // -> 15
  cheb(buf0, S0, 15, buf1, S1, p.cw[1], p.cb[1], lane);            // -> 15
  tconv(buf1, S1, 15, buf0, S0, p.w[3][0], p.w[3][1], p.w[3][2],
        p.bw[3][0], p.bw[3][1], p.bw[3][2], lane);                 // -> 13

  // ---- BN + channel mean + global node reduction ----
  finalStage(buf0, S0, p.bng[1], p.bnb[1], nc, valid, lane, p.S);
}

__global__ void stgcn_zero(float* S) {
  if (threadIdx.x < 13) S[threadIdx.x] = 0.0f;
}

__global__ void stgcn_final(const float* S, const float* lw, const float* lb,
                            float* out) {
  int o = threadIdx.x;
  if (o < 10) {
    const float inv = 1.0f / ((float)N_NODES * 32.0f);
    float acc = lb[o];
#pragma unroll
    for (int t = 0; t < 13; ++t) acc += (S[t] * inv) * lw[o * 13 + t];
    out[o] = acc;
  }
}

extern "C" void kernel_launch(void* const* d_in, const int* in_sizes, int n_in,
                              void* d_out, int out_size, void* d_ws,
                              size_t ws_size, hipStream_t stream) {
  (void)in_sizes; (void)n_in; (void)out_size; (void)ws_size;
  // setup_inputs() order: x, edge_index, edge_weight, params{st1{tc1[3x(w,b)],
  // cheb_w, cheb_b, tc2[3x(w,b)], bn_g, bn_b}, st2{...}, lin_w, lin_b}.
  // edge_index (d_in[1]) / edge_weight (d_in[2]) are unused (ChebConv K=1).
  STGCNParams p;
  p.x = (const float*)d_in[0];
  const int iw[4][3] = {{3, 5, 7}, {11, 13, 15}, {19, 21, 23}, {27, 29, 31}};
  const int ib[4][3] = {{4, 6, 8}, {12, 14, 16}, {20, 22, 24}, {28, 30, 32}};
  for (int l = 0; l < 4; ++l)
    for (int g = 0; g < 3; ++g) {
      p.w[l][g]  = (const float*)d_in[iw[l][g]];
      p.bw[l][g] = (const float*)d_in[ib[l][g]];
    }
  p.cw[0] = (const float*)d_in[9];  p.cb[0] = (const float*)d_in[10];
  p.cw[1] = (const float*)d_in[25]; p.cb[1] = (const float*)d_in[26];
  p.bng[0] = (const float*)d_in[17]; p.bnb[0] = (const float*)d_in[18];
  p.bng[1] = (const float*)d_in[33]; p.bnb[1] = (const float*)d_in[34];
  const float* lw = (const float*)d_in[35];
  const float* lb = (const float*)d_in[36];
  p.S = (float*)d_ws;

  stgcn_zero<<<1, 32, 0, stream>>>(p.S);

  const size_t shmem = (size_t)(WAVES * NPW * (S0 + S1)) * sizeof(_Float16);
  const int grid = (N_NODES + NPB - 1) / NPB;
  stgcn_main<<<grid, WAVES * 32, shmem, stream>>>(p);

  stgcn_final<<<1, 32, 0, stream>>>(p.S, lw, lb, (float*)d_out);
}
